// Classifier_71768903516861
// MI455X (gfx1250) — matrix operand order repaired and checked by reading, not compiled, not run on hardware
//
#include <hip/hip_runtime.h>

typedef _Float16 f16;
typedef __attribute__((ext_vector_type(8)))  _Float16 v8h;
typedef __attribute__((ext_vector_type(16))) _Float16 v16h;
typedef __attribute__((ext_vector_type(8)))  float    v8f;

#define DEV __device__ __forceinline__

DEV v16h cat16(v8h a, v8h b) {
  return __builtin_shufflevector(a, b, 0,1,2,3,4,5,6,7,8,9,10,11,12,13,14,15);
}

DEV v8f wmma16(v16h a, v16h b, v8f c) {
  return __builtin_amdgcn_wmma_f32_16x16x32_f16(false, a, false, b,
                                                (short)0, c, false, false);
}

// ---------------------------------------------------------------------------
// Generic WMMA GEMM: C[M,N] = A[M,K] @ B[K,N] (+bias per row) (opt ReLU)
// A: f16 row-major (M x K). B: f16 "K-major" (K rows of length N).
// Requires M%32==0, N%64==0, K%32==0. One wave -> 32x64 output tile
// (2 M-frags x 4 N-frags = 8 WMMAs per K-step vs 12 b128 loads: 21 FLOP/B).
// A frag (16x32, MxK): lane(m=lane&15,h=lane>>4): K = k0 + h*8+{0..7} and
//                      k0 + 16 + h*8 + {0..7}  (two b128 loads)
// B frag (32x16, KxN): lane(n=lane&15,h=lane>>4): same K order as A (strided loads)
// D: row = v + 8*(lane>>4), col = lane&15
// ---------------------------------------------------------------------------
__global__ __launch_bounds__(256) void k_wmma_gemm(
    const f16* __restrict__ A, const f16* __restrict__ B,
    float* __restrict__ C, const float* __restrict__ bias,
    int M, int N, int K, int do_relu)
{
  const int NT = N >> 6;                       // 64-wide tiles
  const int MT = M >> 5;                       // 32-tall tiles
  const int wave = blockIdx.x * 8 + (threadIdx.x >> 5);
  if (wave >= MT * NT) return;                 // wave-uniform: EXEC stays all-1
  const int lane = threadIdx.x & 31;
  const int mt = wave / NT, nt = wave % NT;
  const int ln = lane & 15, lh = lane >> 4;

  const f16* Ap0 = A + (size_t)(mt * 32 + ln) * K + lh * 8;
  const f16* Ap1 = Ap0 + (size_t)16 * K;
  const f16* Bp  = B + (size_t)(lh * 8) * N + nt * 64 + ln;
  const size_t BkS = (size_t)32 * N;

  v8f zero = {};
  v8f acc[8];
#pragma unroll
  for (int i = 0; i < 8; ++i) acc[i] = zero;

  for (int k0 = 0; k0 < K; k0 += 32) {
    v16h a0 = cat16(*(const v8h*)(Ap0), *(const v8h*)(Ap0 + 16));
    v16h a1 = cat16(*(const v8h*)(Ap1), *(const v8h*)(Ap1 + 16));
    if (k0 + 32 < K) __builtin_prefetch((const void*)(Bp + BkS), 0, 1);
#pragma unroll
    for (int j = 0; j < 4; ++j) {
      v16h bf;
#pragma unroll
      for (int s = 0; s < 8; ++s) {
        bf[s]     = Bp[(size_t)s * N + 16 * j];
        bf[8 + s] = Bp[(size_t)(16 + s) * N + 16 * j];
      }
      acc[j]     = wmma16(a0, bf, acc[j]);
      acc[4 + j] = wmma16(a1, bf, acc[4 + j]);
    }
    Ap0 += 32;
    Ap1 += 32;
    Bp  += BkS;
  }

#pragma unroll
  for (int j = 0; j < 4; ++j) {
    const int col = nt * 64 + 16 * j + ln;
#pragma unroll
    for (int v = 0; v < 8; ++v) {
      const int r0 = mt * 32 + v + 8 * lh;
      const int r1 = r0 + 16;
      const float bv0 = bias ? bias[r0] : 0.0f;
      const float bv1 = bias ? bias[r1] : 0.0f;
      float x0 = acc[j][v] + bv0;
      float x1 = acc[4 + j][v] + bv1;
      if (do_relu) { x0 = fmaxf(x0, 0.0f); x1 = fmaxf(x1, 0.0f); }
      C[(size_t)r0 * N + col] = x0;
      C[(size_t)r1 * N + col] = x1;
    }
  }
}

// ------------------------------ helper kernels ------------------------------

__global__ void k_f32_to_f16(const float* __restrict__ s, f16* __restrict__ d, int n) {
  int i = blockIdx.x * blockDim.x + threadIdx.x;
  if (i < n) d[i] = (f16)s[i];
}

// zero-pad rows: dst[rows x cols] f16, src[src_rows x cols] f32
__global__ void k_pad_rows(const float* __restrict__ s, f16* __restrict__ d,
                           int src_rows, int rows, int cols) {
  int i = blockIdx.x * blockDim.x + threadIdx.x;
  if (i >= rows * cols) return;
  int r = i / cols, c = i % cols;
  d[i] = (f16)((r < src_rows) ? s[r * cols + c] : 0.0f);
}

__global__ void k_pad_bias(const float* __restrict__ s, float* __restrict__ d,
                           int n, int total) {
  int i = blockIdx.x * blockDim.x + threadIdx.x;
  if (i < total) d[i] = (i < n) ? s[i] : 0.0f;
}

// Xpack[c][b*49+hw] = x[b][c][hw]   (K-major operand for the big GEMM)
__global__ void k_pack_x(const float* __restrict__ x, f16* __restrict__ Xp, int total) {
  int i = blockIdx.x * blockDim.x + threadIdx.x;
  if (i >= total) return;
  int c = i / 12544, n = i % 12544;
  int b = n / 49, hw = n % 49;
  Xp[i] = (f16)x[((size_t)b * 2048 + c) * 49 + hw];
}

__global__ void k_mean_pool(const float* __restrict__ x, float* __restrict__ xp, int total) {
  int i = blockIdx.x * blockDim.x + threadIdx.x;
  if (i >= total) return;
  const float* p = x + (size_t)i * 49;
  float s = 0.0f;
  for (int k = 0; k < 49; ++k) s += p[k];
  xp[i] = s * (1.0f / 49.0f);
}

// out[r,c] = sum_k X[r*K+k]*W[c*K+k] (+bias[c]); optional duplicate write
__global__ void k_vgemm(const float* __restrict__ X, const float* __restrict__ W,
                        const float* __restrict__ bias, float* __restrict__ out,
                        float* __restrict__ out2, int R, int C, int K) {
  int i = blockIdx.x * blockDim.x + threadIdx.x;
  if (i >= R * C) return;
  int r = i / C, c = i % C;
  const float* xp = X + (size_t)r * K;
  const float* wp = W + (size_t)c * K;
  float s = bias ? bias[c] : 0.0f;
  for (int k = 0; k < K; ++k) s = fmaf(xp[k], wp[k], s);
  out[i] = s;
  if (out2) out2[i] = s;
}

__global__ void k_softmax_rows(const float* __restrict__ src, float* __restrict__ dst,
                               int rows, int cols) {
  int r = blockIdx.x * blockDim.x + threadIdx.x;
  if (r >= rows) return;
  const float* s = src + (size_t)r * cols;
  float* d = dst + (size_t)r * cols;
  float mx = s[0];
  for (int i = 1; i < cols; ++i) mx = fmaxf(mx, s[i]);
  float sum = 0.0f;
  for (int i = 0; i < cols; ++i) sum += __expf(s[i] - mx);
  float inv = 1.0f / sum;
  for (int i = 0; i < cols; ++i) d[i] = __expf(s[i] - mx) * inv;
}

// Wpr[i,j] = 0.85*edge[i,j] / rowsum(edge)[j]
__global__ void k_wpr(const float* __restrict__ edge, float* __restrict__ Wpr) {
  int i = blockIdx.x * blockDim.x + threadIdx.x;
  if (i >= 87 * 87) return;
  int c = i % 87;
  float s = 0.0f;
  for (int k = 0; k < 87; ++k) s += edge[c * 87 + k];
  Wpr[i] = 0.85f * edge[i] / s;
}

// one block per batch; 100 power iterations entirely in LDS
__global__ __launch_bounds__(128) void k_pagerank(
    const float* __restrict__ Wpr, const float* __restrict__ pre_score,
    float* __restrict__ v_y)
{
  __shared__ float sW[87 * 87];
  __shared__ float sv[87], snv[87], ss[87];
  __shared__ float red_s;
  const int b = blockIdx.x;
  const int t = threadIdx.x;
  for (int i = t; i < 87 * 87; i += 128) sW[i] = Wpr[i];
  if (t < 87) {
    float vi = (t >= 22) ? pre_score[b * 65 + (t - 22)] : 0.0f;
    ss[t] = vi * 87.0f;   // s = v_init * n
    sv[t] = ss[t];        // xv = s
  }
  __syncthreads();
  const float coef = (1.0f - 0.85f) / 87.0f;
  for (int it = 0; it < 100; ++it) {
    if (t == 0) {
      float s = 0.0f;
      for (int i = 0; i < 87; ++i) s += sv[i];
      red_s = s * coef;
    }
    __syncthreads();
    if (t < 87) {
      float acc = ss[t] * red_s;
      const float* wr = &sW[t * 87];
      for (int j = 0; j < 87; ++j) acc = fmaf(wr[j], sv[j], acc);
      snv[t] = acc;
    }
    __syncthreads();
    if (t < 87) sv[t] = snv[t];
    __syncthreads();
  }
  if (t == 0) {
    float s = 0.0f;
    for (int i = 0; i < 87; ++i) s += sv[i] * sv[i];
    red_s = fmaxf(sqrtf(s), 1e-12f);
  }
  __syncthreads();
  if (t < 87) {
    float vin = (t >= 22) ? pre_score[b * 65 + (t - 22)] : 0.0f;
    v_y[b * 87 + t] = vin + sv[t] / red_s;
  }
}

// concat rows 0..127 : node[b,k,m] = node_p[k,m] * v_y[(b&1)*128+k, m]
__global__ void k_node(const float* __restrict__ node_p, const float* __restrict__ v_y,
                       f16* __restrict__ concat) {
  int i = blockIdx.x * blockDim.x + threadIdx.x;
  if (i >= 128 * 22272) return;
  int k = i / 22272, rem = i % 22272;
  int b = rem / 87, m = rem % 87;
  float v = node_p[k * 87 + m] * v_y[(((b & 1) << 7) + k) * 87 + m];
  concat[i] = (f16)v;
}

// concat rows 128..255 : m1[b,c,m] = relu(sum_hw ps[b,c,hw]*a[b,hw,m])
__global__ void k_m1(const float* __restrict__ ps, const float* __restrict__ amat,
                     f16* __restrict__ concat) {
  int i = blockIdx.x * blockDim.x + threadIdx.x;
  if (i >= 128 * 22272) return;
  int c = i / 22272, rem = i % 22272;
  int b = rem / 87, m = rem % 87;
  const float* pp = ps   + (size_t)c * 12544 + b * 49;
  const float* ap = amat + (size_t)m * 12544 + b * 49;
  float s = 0.0f;
  for (int hw = 0; hw < 49; ++hw) s = fmaf(pp[hw], ap[hw], s);
  concat[(size_t)(128 + c) * 22272 + rem] = (f16)fmaxf(s, 0.0f);
}

// gA[(b*128+c)][m<96] = g[c][b*87+m]  (A operand for the m2 GEMM, zero-padded)
__global__ void k_pack_g(const float* __restrict__ g, f16* __restrict__ gA) {
  int i = blockIdx.x * blockDim.x + threadIdx.x;
  if (i >= 32768 * 96) return;
  int row = i / 96, m = i % 96;
  int b = row >> 7, c = row & 127;
  float v = (m < 87) ? g[(size_t)c * 22272 + b * 87 + m] : 0.0f;
  gA[i] = (f16)v;
}

// edge_norm zero-padded to 96x128, f16 (B operand): Dinv[m]*edge[m,n]*Dinv[n]
__global__ void k_edge_norm(const float* __restrict__ edge, f16* __restrict__ out) {
  int i = blockIdx.x * blockDim.x + threadIdx.x;
  if (i >= 96 * 128) return;
  int r = i / 128, c = i % 128;
  float v = 0.0f;
  if (r < 87 && c < 87) {
    float sr = 0.0f, sc = 0.0f;
    for (int k = 0; k < 87; ++k) { sr += edge[k * 87 + r]; sc += edge[k * 87 + c]; }
    v = edge[r * 87 + c] * rsqrtf(sr) * rsqrtf(sc);
  }
  out[i] = (f16)v;
}

// m2T[c][b*87+m] = m2[(b*128+c)*128 + m]  (K-major operand for sp GEMM)
__global__ void k_pack_m2T(const float* __restrict__ m2, f16* __restrict__ m2T) {
  int i = blockIdx.x * blockDim.x + threadIdx.x;
  if (i >= 128 * 22272) return;
  int c = i / 22272, rem = i % 22272;
  int b = rem / 87, m = rem % 87;
  m2T[i] = (f16)m2[((size_t)(b * 128 + c)) * 128 + m];
}

__global__ void k_tnode(const float* __restrict__ Ws, const float* __restrict__ m2,
                        float* __restrict__ tnode) {
  int i = blockIdx.x * blockDim.x + threadIdx.x;
  if (i >= 22272) return;
  int b = i / 87, m = i % 87;
  float s = 0.0f;
  for (int c = 0; c < 128; ++c)
    s = fmaf(Ws[c], m2[((size_t)(b * 128 + c)) * 128 + m], s);
  tnode[i] = s;
}

__global__ void k_tpix(const float* __restrict__ Ws, const float* __restrict__ xl,
                       float* __restrict__ tpix) {
  int i = blockIdx.x * blockDim.x + threadIdx.x;
  if (i >= 12544) return;
  float s = 0.0f;
  for (int c = 0; c < 128; ++c) s = fmaf(Ws[128 + c], xl[(size_t)c * 12544 + i], s);
  tpix[i] = s;
}

// softmax over m of (t_node[b,m] + t_pix[b,p] + bs)  -> satt[b][m][p]
__global__ void k_satt(const float* __restrict__ tnode, const float* __restrict__ tpix,
                       const float* __restrict__ bs, float* __restrict__ satt) {
  int i = blockIdx.x * blockDim.x + threadIdx.x;
  if (i >= 256 * 49) return;
  int b = i / 49, p = i % 49;
  float tp = tpix[i] + bs[0];
  const float* tn = tnode + b * 87;
  float mx = -1e30f;
  for (int m = 0; m < 87; ++m) mx = fmaxf(mx, tn[m] + tp);
  float sum = 0.0f;
  for (int m = 0; m < 87; ++m) sum += __expf(tn[m] + tp - mx);
  float inv = 1.0f / sum;
  float* out = satt + ((size_t)b * 87) * 49 + p;
  for (int m = 0; m < 87; ++m) out[(size_t)m * 49] = __expf(tn[m] + tp - mx) * inv;
}

// m3pool[b,c] = mean_p relu(sum_m sp[b,c,m]*satt[b,m,p]);  block per (b,c)
__global__ __launch_bounds__(64) void k_m3pool(const float* __restrict__ sp,
                                               const float* __restrict__ satt,
                                               float* __restrict__ m3pool) {
  __shared__ float red[64];
  int blk = blockIdx.x;
  int b = blk >> 7, c = blk & 127;
  int t = threadIdx.x;
  float v = 0.0f;
  if (t < 49) {
    const float* spp = sp + (size_t)c * 22272 + b * 87;
    const float* sa  = satt + ((size_t)b * 87) * 49 + t;
    float s = 0.0f;
    for (int m = 0; m < 87; ++m) s = fmaf(spp[m], sa[(size_t)m * 49], s);
    v = fmaxf(s, 0.0f);
  }
  red[t] = v;
  __syncthreads();
  for (int o = 32; o > 0; o >>= 1) { if (t < o) red[t] += red[t + o]; __syncthreads(); }
  if (t == 0) m3pool[blk] = red[0] * (1.0f / 49.0f);
}

// features[b,j] = features_p[b,j] + sum_c m3pool[b,c]*Wcomb[j,c] + bconst[j]
__global__ void k_tail(const float* __restrict__ m3pool, const float* __restrict__ Wcomb,
                       const float* __restrict__ bconst, const float* __restrict__ featp,
                       float* __restrict__ out) {
  int i = blockIdx.x * blockDim.x + threadIdx.x;
  if (i >= 256 * 256) return;
  int b = i / 256, j = i % 256;
  const float* mp = m3pool + b * 128;
  const float* wc = Wcomb + (size_t)j * 128;
  float s = bconst[j];
  for (int c = 0; c < 128; ++c) s = fmaf(mp[c], wc[c], s);
  out[i] = featp[i] + s;
}

// ------------------------------- host driver --------------------------------

extern "C" void kernel_launch(void* const* d_in, const int* in_sizes, int n_in,
                              void* d_out, int out_size, void* d_ws, size_t ws_size,
                              hipStream_t stream) {
  (void)in_sizes; (void)n_in; (void)out_size; (void)ws_size;
  const float* x      = (const float*)d_in[0];
  const float* Wb     = (const float*)d_in[1];
  const float* bb     = (const float*)d_in[2];
  const float* Wfc    = (const float*)d_in[3];
  const float* Wfc2   = (const float*)d_in[4];
  const float* Wo1    = (const float*)d_in[5];
  const float* bo1    = (const float*)d_in[6];
  const float* Wo2    = (const float*)d_in[7];
  const float* bo2    = (const float*)d_in[8];
  const float* Wa     = (const float*)d_in[9];
  const float* ba     = (const float*)d_in[10];
  const float* Wps    = (const float*)d_in[11];
  const float* bps    = (const float*)d_in[12];
  const float* node_p = (const float*)d_in[13];
  const float* Wg     = (const float*)d_in[14];
  const float* bg     = (const float*)d_in[15];
  const float* Wsp    = (const float*)d_in[16];
  const float* bsp    = (const float*)d_in[17];
  const float* Ws     = (const float*)d_in[18];
  const float* bs     = (const float*)d_in[19];
  const float* Wox1   = (const float*)d_in[20];
  const float* box1   = (const float*)d_in[21];
  const float* Wox2   = (const float*)d_in[22];
  const float* box2   = (const float*)d_in[23];
  const float* edge   = (const float*)d_in[24];
  float* dout = (float*)d_out;

  const int O_FEAT = 0, O_OUT2 = 65536, O_SOFT = 82176, O_PRE = 98816, O_FEATP = 115456;

  size_t off = 0;
  auto alloc = [&](size_t bytes) -> void* {
    off = (off + 255) & ~(size_t)255;
    void* p = (char*)d_ws + off;
    off += bytes;
    return p;
  };
  auto F  = [&](size_t n) { return (float*)alloc(n * 4); };
  auto H  = [&](size_t n) { return (f16*)alloc(n * 2); };

  f16*   Xpack  = H((size_t)2048 * 12544);
  f16*   Wo1h   = H((size_t)512 * 2048);
  f16*   Wo2h   = H(128 * 512);
  float* Wfoldf = F(128 * 2048);
  f16*   Wfoldh = H(128 * 2048);
  float* bfold  = F(128);
  float* xl     = F((size_t)128 * 12544);
  f16*   xlh    = H((size_t)128 * 12544);
  f16*   Wah    = H(96 * 128);
  float* bap    = F(96);
  float* amat   = F((size_t)96 * 12544);
  f16*   Wpsh   = H(128 * 128);
  float* ps     = F((size_t)128 * 12544);
  float* xp     = F(256 * 2048);
  float* featp  = F(256 * 256);
  float* preout = F(256 * 65);
  float* prescr = F(256 * 65);
  float* Wpr    = F(87 * 87);
  float* vy     = F(256 * 87);
  f16*   concat = H((size_t)256 * 22272);
  f16*   Wgh    = H(128 * 256);
  float* gbuf   = F((size_t)128 * 22272);
  f16*   gA     = H((size_t)32768 * 96);
  f16*   edgeh  = H(96 * 128);
  float* m2buf  = F((size_t)32768 * 128);
  f16*   m2T    = H((size_t)128 * 22272);
  f16*   Wsph   = H(128 * 128);
  float* spbuf  = F((size_t)128 * 22272);
  float* tnode  = F(22272);
  float* tpix   = F(12544);
  float* satt   = F((size_t)256 * 87 * 49);
  float* m3pool = F(32768);
  f16*   Wox1h  = H(512 * 128);
  f16*   Wox2h  = H((size_t)2048 * 512);
  float* Wt1f   = F((size_t)2048 * 128);
  f16*   Wt1h   = H((size_t)2048 * 128);
  f16*   Wbh    = H((size_t)256 * 2048);
  float* Wcomb  = F(256 * 128);
  float* v1     = F(2048);
  float* bconst = F(256);

  auto blocks = [](long n) { return dim3((unsigned)((n + 255) / 256)); };
  auto conv = [&](const float* s, f16* d, long n) {
    k_f32_to_f16<<<blocks(n), 256, 0, stream>>>(s, d, (int)n);
  };
  auto gemm = [&](const f16* A, const f16* Bm, float* Cm, const float* bias,
                  int M, int N, int K, int relu) {
    int tiles = (M / 32) * (N / 64);
    k_wmma_gemm<<<dim3((unsigned)((tiles + 7) / 8)), 256, 0, stream>>>(
        A, Bm, Cm, bias, M, N, K, relu);
  };

  // ---- input packing / weight conversion ----
  k_pack_x<<<blocks((long)2048 * 12544), 256, 0, stream>>>(x, Xpack, 2048 * 12544);
  k_mean_pool<<<blocks(256 * 2048), 256, 0, stream>>>(x, xp, 256 * 2048);
  conv(Wo1, Wo1h, 512 * 2048);
  conv(Wo2, Wo2h, 128 * 512);
  conv(Wps, Wpsh, 128 * 128);
  conv(Wg,  Wgh,  128 * 256);
  conv(Wsp, Wsph, 128 * 128);
  conv(Wox1, Wox1h, 512 * 128);
  conv(Wox2, Wox2h, 2048 * 512);
  conv(Wb, Wbh, 256 * 2048);
  k_pad_rows<<<blocks(96 * 128), 256, 0, stream>>>(Wa, Wah, 87, 96, 128);
  k_pad_bias<<<blocks(96), 256, 0, stream>>>(ba, bap, 87, 96);

  // ---- classifier head + pagerank ----
  k_vgemm<<<blocks(256 * 256), 256, 0, stream>>>(xp, Wb, bb, featp, dout + O_FEATP,
                                                 256, 256, 2048);
  k_vgemm<<<blocks(256 * 65), 256, 0, stream>>>(featp, Wfc, nullptr, preout,
                                                dout + O_PRE, 256, 65, 256);
  k_softmax_rows<<<blocks(256), 256, 0, stream>>>(preout, prescr, 256, 65);
  k_wpr<<<blocks(87 * 87), 256, 0, stream>>>(edge, Wpr);
  k_pagerank<<<dim3(256), dim3(128), 0, stream>>>(Wpr, prescr, vy);

  // ---- folded front convs: x_l = (Wo2@Wo1)@x + (Wo2@bo1+bo2) ----
  gemm(Wo2h, Wo1h, Wfoldf, nullptr, 128, 2048, 512, 0);
  conv(Wfoldf, Wfoldh, 128 * 2048);
  k_vgemm<<<blocks(128), 256, 0, stream>>>(bo1, Wo2, bo2, bfold, nullptr, 1, 128, 512);
  gemm(Wfoldh, Xpack, xl, bfold, 128, 12544, 2048, 0);
  conv(xl, xlh, (long)128 * 12544);

  // ---- a / ps heads ----
  gemm(Wah,  xlh, amat, bap, 96, 12544, 128, 0);
  gemm(Wpsh, xlh, ps,   bps, 128, 12544, 128, 0);

  // ---- graph branch ----
  k_node<<<blocks((long)128 * 22272), 256, 0, stream>>>(node_p, vy, concat);
  k_m1<<<blocks((long)128 * 22272), 256, 0, stream>>>(ps, amat, concat);
  gemm(Wgh, concat, gbuf, bg, 128, 22272, 256, 0);
  k_pack_g<<<blocks((long)32768 * 96), 256, 0, stream>>>(gbuf, gA);
  k_edge_norm<<<blocks(96 * 128), 256, 0, stream>>>(edge, edgeh);
  gemm(gA, edgeh, m2buf, nullptr, 32768, 128, 96, 1);
  k_pack_m2T<<<blocks((long)128 * 22272), 256, 0, stream>>>(m2buf, m2T);
  k_tnode<<<blocks(22272), 256, 0, stream>>>(Ws, m2buf, tnode);
  gemm(Wsph, m2T, spbuf, bsp, 128, 22272, 128, 0);

  // ---- attention + pooled m3 ----
  k_tpix<<<blocks(12544), 256, 0, stream>>>(Ws, xl, tpix);
  k_satt<<<blocks(256 * 49), 256, 0, stream>>>(tnode, tpix, bs, satt);
  k_m3pool<<<dim3(32768), dim3(64), 0, stream>>>(spbuf, satt, m3pool);

  // ---- folded tail: Wcomb = Wb @ (Wox2@Wox1), bconst = Wb@(Wox2@box1+box2)+bb
  gemm(Wox2h, Wox1h, Wt1f, nullptr, 2048, 128, 512, 0);
  conv(Wt1f, Wt1h, (long)2048 * 128);
  gemm(Wbh, Wt1h, Wcomb, nullptr, 256, 128, 2048, 0);
  k_vgemm<<<blocks(2048), 256, 0, stream>>>(box1, Wox2, box2, v1, nullptr, 1, 2048, 512);
  k_vgemm<<<blocks(256), 256, 0, stream>>>(v1, Wb, bb, bconst, nullptr, 1, 256, 2048);

  // ---- outputs ----
  k_tail<<<blocks(256 * 256), 256, 0, stream>>>(m3pool, Wcomb, bconst, featp,
                                                dout + O_FEAT);
  k_vgemm<<<blocks(256 * 65), 256, 0, stream>>>(dout + O_FEAT, Wfc2, nullptr,
                                                dout + O_OUT2, nullptr, 256, 65, 256);
  k_softmax_rows<<<blocks(256), 256, 0, stream>>>(dout + O_OUT2, dout + O_SOFT, 256, 65);
}
